// BaseAC_22505628631094
// MI455X (gfx1250) — compile-verified, matrix operand order checked by reading, not run on hardware
//
#include <hip/hip_runtime.h>
#include <hip/hip_bf16.h>

// ---------------------------------------------------------------------------
// MI455X (gfx1250) fused GAT-autoencoder:
//   K1: encoder GEMM (feature_src @ fc_W^T + fc_b) -> transformed (f16 RM + f16 T)
//   K2: decoder GEMM (transformed @ dec_W^T + dec_b) -> feature_hat (f32 out)
//   K3: per-head projections h_src = emb_src@W,  g_dst = (emb_dest@W)@W2 (f16)
//   K4: fused flash-attention: elu(g_dst @ h_src^T), bias mask via TDM-staged
//       LDS tiles, online softmax, P @ transformed, mean over heads.
// Matrix math: v_wmma_f32_16x16x32_f16 (wave32, f32 accumulate).
// Bias stream (256 MB, the dominant HBM traffic): double-buffered
// tensor_load_to_lds DMA overlapped with WMMA compute, s_wait_tensorcnt sync.
// ---------------------------------------------------------------------------

typedef _Float16 h16;
typedef __attribute__((ext_vector_type(16))) _Float16 v16h;
typedef __attribute__((ext_vector_type(8)))  float    v8f;
typedef __attribute__((ext_vector_type(4)))  unsigned int u32x4;
typedef __attribute__((ext_vector_type(8)))  int      i32x8;
typedef __attribute__((ext_vector_type(4)))  int      i32x4;

#define N_DEST 8192
#define N_SRC  8192
#define FEAT   1024
#define TFD    256
#define EMB    64
#define HID    128
#define NEG_INF_REF (-9.0e15f)
#define NEG_BIG     (-3.0e38f)

#if defined(__AMDGCN__) && __has_builtin(__builtin_amdgcn_tensor_load_to_lds) && \
    __has_builtin(__builtin_amdgcn_s_wait_tensorcnt)
#define HAVE_TDM 1
#endif

static __device__ __forceinline__ v8f wmma16(v16h a, v16h b, v8f c) {
  return __builtin_amdgcn_wmma_f32_16x16x32_f16(
      /*neg_a=*/false, a, /*neg_b=*/false, b,
      /*c_mod=*/(short)0, c, /*reuse_a=*/false, /*reuse_b=*/false);
}

#ifdef HAVE_TDM
// 2D TDM descriptor: tile = 128 (contig, f32) x 16 rows out of an
// 8192 x 8192 f32 tensor with row stride 8192 elements; dest = LDS offset.
static __device__ __forceinline__ void tdm_load_bias_tile(const float* gptr,
                                                          unsigned lds_off) {
  const unsigned long long ga = (unsigned long long)(size_t)gptr;
  u32x4 g0;
  g0[0] = 1u;                                        // count=1, user D#
  g0[1] = lds_off;                                   // lds_addr (bytes)
  g0[2] = (unsigned)(ga & 0xffffffffu);              // global_addr[31:0]
  g0[3] = (unsigned)((ga >> 32) & 0x1ffffffu) | (2u << 30); // addr[56:32], type=2
  i32x8 g1;
  g1[0] = (2 << 16);                                 // data_size=4B, mask=0
  g1[1] = (int)((N_SRC & 0xffff) << 16);             // tensor_dim0[15:0]
  g1[2] = (int)(((N_SRC >> 16) & 0xffff) | ((N_DEST & 0xffff) << 16));
  g1[3] = (int)(((N_DEST >> 16) & 0xffff) | (128 << 16)); // tile_dim0=128
  g1[4] = 16;                                        // tile_dim1=16, tile_dim2=0
  g1[5] = N_SRC;                                     // tensor_dim0_stride lo
  g1[6] = 0;
  g1[7] = 0;
  i32x4 z4 = {0, 0, 0, 0};
#if __clang_major__ >= 23
  i32x8 z8 = {0, 0, 0, 0, 0, 0, 0, 0};
  __builtin_amdgcn_tensor_load_to_lds(g0, g1, z4, z4, z8, 0);
#else
  __builtin_amdgcn_tensor_load_to_lds(g0, g1, z4, z4, 0);
#endif
}
#endif  // HAVE_TDM

// ------------------------------ K1: encoder --------------------------------
// transformed[m][n] = sum_k feature_src[m][k] * fc_W[n][k] + fc_b[n]
__global__ __launch_bounds__(256)
void encoder_kernel(const float* __restrict__ feat, const float* __restrict__ fcW,
                    const float* __restrict__ fcb,
                    h16* __restrict__ T, h16* __restrict__ Tt) {
  const int lane = threadIdx.x & 31, wave = threadIdx.x >> 5;
  const int lrow = lane & 15, lhi = lane >> 4;
  const int m0 = blockIdx.x * 16;
  const int n0 = wave * 32;

  v8f acc0 = {}, acc1 = {};
  for (int k0 = 0; k0 < FEAT; k0 += 32) {
    v16h a, b0, b1;
    const float* ap  = feat + (size_t)(m0 + lrow) * FEAT + k0 + lhi * 16;
    const float* bp0 = fcW  + (size_t)(n0 + lrow) * FEAT + k0 + lhi * 16;
    const float* bp1 = fcW  + (size_t)(n0 + 16 + lrow) * FEAT + k0 + lhi * 16;
#pragma unroll
    for (int j = 0; j < 16; ++j) {
      a[j]  = (_Float16)ap[j];
      b0[j] = (_Float16)bp0[j];
      b1[j] = (_Float16)bp1[j];
    }
    acc0 = wmma16(a, b0, acc0);
    acc1 = wmma16(a, b1, acc1);
  }
#pragma unroll
  for (int r = 0; r < 8; ++r) {
    const int m = m0 + r + 8 * lhi;
    const int n = n0 + lrow;
    const float v0 = acc0[r] + fcb[n];
    const float v1 = acc1[r] + fcb[n + 16];
    T[(size_t)m * TFD + n]           = (_Float16)v0;
    T[(size_t)m * TFD + n + 16]      = (_Float16)v1;
    Tt[(size_t)n * N_SRC + m]        = (_Float16)v0;
    Tt[(size_t)(n + 16) * N_SRC + m] = (_Float16)v1;
  }
}

// ------------------------------ K2: decoder --------------------------------
__global__ __launch_bounds__(256)
void decoder_kernel(const h16* __restrict__ T, const float* __restrict__ decW,
                    const float* __restrict__ decb, float* __restrict__ out_hat) {
  const int lane = threadIdx.x & 31, wave = threadIdx.x >> 5;
  const int lrow = lane & 15, lhi = lane >> 4;
  const int m0 = blockIdx.x * 16;

  for (int nt = 0; nt < 8; ++nt) {
    const int n0 = (wave * 8 + nt) * 16;
    v8f acc = {};
    for (int k0 = 0; k0 < TFD; k0 += 32) {
      v16h a, b;
      const h16*   ap = T    + (size_t)(m0 + lrow) * TFD + k0 + lhi * 16;
      const float* bp = decW + (size_t)(n0 + lrow) * TFD + k0 + lhi * 16;
#pragma unroll
      for (int j = 0; j < 16; ++j) {
        a[j] = ap[j];
        b[j] = (_Float16)bp[j];
      }
      acc = wmma16(a, b, acc);
    }
#pragma unroll
    for (int r = 0; r < 8; ++r) {
      const int m = m0 + r + 8 * lhi;
      const int n = n0 + lrow;
      out_hat[(size_t)m * FEAT + n] = acc[r] + decb[n];
    }
  }
}

// --------------------------- K3: projections -------------------------------
__global__ __launch_bounds__(256)
void proj_kernel(const float* __restrict__ emb_dest, const float* __restrict__ emb_src,
                 const float* __restrict__ attW, const float* __restrict__ attW2,
                 h16* __restrict__ hsrc, h16* __restrict__ gdst) {
  const int lane = threadIdx.x & 31, wave = threadIdx.x >> 5;
  const int lrow = lane & 15, lhi = lane >> 4;
  const int m0   = blockIdx.x * 16;
  const int head = blockIdx.y;
  const int isdst = blockIdx.z;
  const float* emb = isdst ? emb_dest : emb_src;
  const int n0 = wave * 16;

  v8f acc = {};
  for (int k0 = 0; k0 < EMB; k0 += 32) {
    v16h a, b;
    const float* ap = emb + (size_t)(m0 + lrow) * EMB + k0 + lhi * 16;
#pragma unroll
    for (int j = 0; j < 16; ++j) {
      a[j] = (_Float16)ap[j];
      b[j] = (_Float16)attW[(size_t)head * EMB * HID + (size_t)(k0 + lhi * 16 + j) * HID + n0 + lrow];
    }
    acc = wmma16(a, b, acc);
  }

  if (!isdst) {
#pragma unroll
    for (int r = 0; r < 8; ++r) {
      const int m = m0 + r + 8 * lhi;
      hsrc[(size_t)head * N_SRC * HID + (size_t)m * HID + n0 + lrow] = (_Float16)acc[r];
    }
  } else {
    __shared__ _Float16 hl[16][HID + 8];
#pragma unroll
    for (int r = 0; r < 8; ++r) {
      hl[r + 8 * lhi][n0 + lrow] = (_Float16)acc[r];
    }
    __syncthreads();
    v8f g = {};
    for (int k0 = 0; k0 < HID; k0 += 32) {
      v16h a2, b2;
#pragma unroll
      for (int j = 0; j < 16; ++j) {
        a2[j] = hl[lrow][k0 + lhi * 16 + j];
        b2[j] = (_Float16)attW2[(size_t)head * HID * HID + (size_t)(k0 + lhi * 16 + j) * HID + n0 + lrow];
      }
      g = wmma16(a2, b2, g);
    }
#pragma unroll
    for (int r = 0; r < 8; ++r) {
      const int m = m0 + r + 8 * lhi;
      gdst[(size_t)head * N_DEST * HID + (size_t)m * HID + n0 + lrow] = (_Float16)g[r];
    }
  }
}

// --------------------------- K4: fused attention ---------------------------
// Per block: 16 dest rows; per head: online softmax over 64 chunks of 128 src.
// Scores computed transposed (S[src][dest]) so the src-softmax reduction is
// per-lane + shfl_xor(16) + small LDS combine. The bias mask tile
// (16 dest x 128 src f32) is DMA'd by the Tensor Data Mover into a
// double-buffered LDS staging area, overlapped with the WMMA work.
__global__ __launch_bounds__(256)
void attn_kernel(const float* __restrict__ bias, const h16* __restrict__ hsrc,
                 const h16* __restrict__ gdst, const h16* __restrict__ Tt,
                 float* __restrict__ out_re) {
  const int lane = threadIdx.x & 31, wave = threadIdx.x >> 5;
  const int lrow = lane & 15, lhi = lane >> 4;
  const int d0 = blockIdx.x * 16;

  __shared__ float sM[16], sL[16], sA[16];
  __shared__ float sPM[8][16], sPS[8][16];
  __shared__ _Float16 sP[16][136];
#ifdef HAVE_TDM
  __shared__ float sB[2][16][128];    // TDM-staged bias tiles (double buffer)
#endif

  v8f R0 = {}, R1 = {};
  const float* biasBase = bias + (size_t)d0 * N_SRC;

  for (int h = 0; h < 2; ++h) {
    if (threadIdx.x < 16) { sM[threadIdx.x] = NEG_BIG; sL[threadIdx.x] = 0.f; }
    __syncthreads();

#ifdef HAVE_TDM
    if (wave == 0) {   // prime the double buffer: chunks 0 and 1 in flight
      tdm_load_bias_tile(biasBase + 0,   (unsigned)(size_t)&sB[0][0][0]);
      tdm_load_bias_tile(biasBase + 128, (unsigned)(size_t)&sB[1][0][0]);
    }
#endif

    // B-operand (constant over src loop): g_dst tile transposed, 128x16 f16
    v16h bg[4];
#pragma unroll
    for (int s = 0; s < 4; ++s) {
      const h16* gp = gdst + (size_t)h * N_DEST * HID + (size_t)(d0 + lrow) * HID + s * 32 + lhi * 16;
#pragma unroll
      for (int j = 0; j < 16; ++j) bg[s][j] = gp[j];
    }

    v8f acc0 = {}, acc1 = {};
    for (int c = 0; c < N_SRC / 128; ++c) {
      const int src0 = c * 128;
      const int s0   = src0 + wave * 16;

#ifdef HAVE_TDM
      if (wave == 0) {                       // this chunk's tile must be done;
        if (c < 63) __builtin_amdgcn_s_wait_tensorcnt(1);  // next may be in flight
        else        __builtin_amdgcn_s_wait_tensorcnt(0);
      }
      __syncthreads();                       // hand staged tile to all waves
#endif

      // --- scores S[src (16 per wave)][dest 16] = h_src @ g_dst^T ---
      v8f sc = {};
#pragma unroll
      for (int s = 0; s < 4; ++s) {
        v16h a;
        const h16* ap = hsrc + (size_t)h * N_SRC * HID + (size_t)(s0 + lrow) * HID + s * 32 + lhi * 16;
#pragma unroll
        for (int j = 0; j < 16; ++j) a[j] = ap[j];
        sc = wmma16(a, bg[s], sc);
      }

      // --- ELU + sparse-mask; per-lane max over this wave's 16 src rows ---
      float msk[8];
#ifdef HAVE_TDM
      const float* bp = &sB[c & 1][lrow][wave * 16 + 8 * lhi];
#else
      const float* bp = bias + (size_t)(d0 + lrow) * N_SRC + s0 + 8 * lhi;
#endif
      float pm = NEG_BIG;
#pragma unroll
      for (int r = 0; r < 8; ++r) {
        const float x = sc[r];
        const float e = x > 0.f ? x : (__expf(x) - 1.f);
        const float v = (bp[r] > 0.f) ? e : NEG_INF_REF;
        msk[r] = v;
        pm = fmaxf(pm, v);
      }
      pm = fmaxf(pm, __shfl_xor(pm, 16, 32));
      if (lhi == 0) sPM[wave][lrow] = pm;
      __syncthreads();

      // --- running max update (one thread per dest column) ---
      if (threadIdx.x < 16) {
        float cm = sPM[0][threadIdx.x];
#pragma unroll
        for (int w = 1; w < 8; ++w) cm = fmaxf(cm, sPM[w][threadIdx.x]);
        const float mOld = sM[threadIdx.x];
        const float mNew = fmaxf(mOld, cm);
        sA[threadIdx.x] = __expf(mOld - mNew);
        sM[threadIdx.x] = mNew;
      }
      __syncthreads();

      // --- probabilities (f16 to LDS) + partial row sums ---
      const float mNew = sM[lrow];
      float ps = 0.f;
#pragma unroll
      for (int r = 0; r < 8; ++r) {
        const float p = __expf(msk[r] - mNew);
        ps += p;
        sP[lrow][wave * 16 + 8 * lhi + r] = (_Float16)p;
      }
      ps += __shfl_xor(ps, 16, 32);
      if (lhi == 0) sPS[wave][lrow] = ps;
      __syncthreads();

      if (threadIdx.x < 16) {
        float l = sL[threadIdx.x] * sA[threadIdx.x];
#pragma unroll
        for (int w = 0; w < 8; ++w) l += sPS[w][threadIdx.x];
        sL[threadIdx.x] = l;
      }

      // --- rescale accumulator, then aggregate P^T @ transformed ---
      float al[8];
#pragma unroll
      for (int r = 0; r < 8; ++r) al[r] = sA[r + 8 * lhi];
#pragma unroll
      for (int r = 0; r < 8; ++r) { acc0[r] *= al[r]; acc1[r] *= al[r]; }

      const int f0 = wave * 32;
#pragma unroll
      for (int s = 0; s < 4; ++s) {
        v16h ap, bt0, bt1;
#pragma unroll
        for (int j = 0; j < 16; ++j) ap[j] = sP[lrow][s * 32 + lhi * 16 + j];
        const h16* tp0 = Tt + (size_t)(f0 + lrow) * N_SRC + src0 + s * 32 + lhi * 16;
        const h16* tp1 = Tt + (size_t)(f0 + 16 + lrow) * N_SRC + src0 + s * 32 + lhi * 16;
#pragma unroll
        for (int j = 0; j < 16; ++j) { bt0[j] = tp0[j]; bt1[j] = tp1[j]; }
        acc0 = wmma16(ap, bt0, acc0);
        acc1 = wmma16(ap, bt1, acc1);
      }
      __syncthreads();  // guards sP/sPM/sB reuse next chunk and sL for finalize

#ifdef HAVE_TDM
      // all reads of sB[c&1] are done (barrier above): refill it for chunk c+2
      if (wave == 0 && c + 2 < N_SRC / 128) {
        tdm_load_bias_tile(biasBase + (size_t)(c + 2) * 128,
                           (unsigned)(size_t)&sB[c & 1][0][0]);
      }
#endif
    }

    // --- finalize head: mean over heads folded in (0.5 / L) ---
    float linv[8];
#pragma unroll
    for (int r = 0; r < 8; ++r) linv[r] = 0.5f / sL[r + 8 * lhi];
#pragma unroll
    for (int r = 0; r < 8; ++r) { R0[r] += acc0[r] * linv[r]; R1[r] += acc1[r] * linv[r]; }
    __syncthreads();
  }

  const int f0 = wave * 32;
#pragma unroll
  for (int r = 0; r < 8; ++r) {
    const int m = d0 + r + 8 * lhi;
    out_re[(size_t)m * TFD + f0 + lrow]      = R0[r];
    out_re[(size_t)m * TFD + f0 + 16 + lrow] = R1[r];
  }
}

// ------------------------------ launcher -----------------------------------
extern "C" void kernel_launch(void* const* d_in, const int* in_sizes, int n_in,
                              void* d_out, int out_size, void* d_ws, size_t ws_size,
                              hipStream_t stream) {
  const float* bias     = (const float*)d_in[0];
  const float* emb_dest = (const float*)d_in[1];
  const float* emb_src  = (const float*)d_in[2];
  const float* feat     = (const float*)d_in[3];
  const float* fcW      = (const float*)d_in[4];
  const float* fcb      = (const float*)d_in[5];
  const float* decW     = (const float*)d_in[6];
  const float* decb     = (const float*)d_in[7];
  const float* attW     = (const float*)d_in[8];
  const float* attW2    = (const float*)d_in[9];

  // workspace layout (f16): 16 MB total
  h16* ws_T    = (h16*)d_ws;                        // [8192][256]  transformed
  h16* ws_Tt   = ws_T  + (size_t)N_SRC * TFD;       // [256][8192]  transformed^T
  h16* ws_hsrc = ws_Tt + (size_t)TFD * N_SRC;       // [2][8192][128]
  h16* ws_gdst = ws_hsrc + (size_t)2 * N_SRC * HID; // [2][8192][128]

  float* out_re  = (float*)d_out;                   // [8192][256]
  float* out_hat = out_re + (size_t)N_DEST * TFD;   // [8192][1024]

  encoder_kernel<<<N_SRC / 16, 256, 0, stream>>>(feat, fcW, fcb, ws_T, ws_Tt);
  proj_kernel<<<dim3(N_SRC / 16, 2, 2), 256, 0, stream>>>(emb_dest, emb_src, attW, attW2,
                                                          ws_hsrc, ws_gdst);
  decoder_kernel<<<N_SRC / 16, 256, 0, stream>>>(ws_T, decW, decb, out_hat);
  attn_kernel<<<N_DEST / 16, 256, 0, stream>>>(bias, ws_hsrc, ws_gdst, ws_Tt, out_re);
}
